// DistributedMoELayer_89721866814266
// MI455X (gfx1250) — matrix operand order repaired and checked by reading, not compiled, not run on hardware
//
#include <hip/hip_runtime.h>
#include <cstdint>

// ---------------- problem constants ----------------
#define NE     8
#define TOPK   2
#define HID    1024
#define INTER  4096
#define TOK    8192          // BATCH*SEQ
#define PADT   16896         // sum of per-expert counts padded to 64

typedef unsigned int u32;

// ---------------- workspace layout (bytes) ----------------
#define OFF_COUNTS  0                       // 8 * i32
#define OFF_CURS    32                      // 8 * i32
#define OFF_PADOFF  64                      // 16 * i32
#define OFF_TEI     128                     // 2*TOK i32      -> 65664
#define OFF_TWF     65664                   // 2*TOK f32      -> 131200
#define OFF_ROWS    131200                  // PADT  i32      -> 198784
#define OFF_WTS     198784                  // PADT  f32      -> 266368
#define OFF_PROBS   266368                  // TOK*8 f32      -> 528512
#define OFF_LSE     528512                  // TOK   f32      -> 561280
#define OFF_XB      561280                  // TOK*HID bf16   -> 17338496
#define OFF_HACT    17338496                // PADT*INTER bf16-> 155750528

// ---------------- types ----------------
typedef __bf16 bf16x16 __attribute__((ext_vector_type(16)));
typedef float  f32x8   __attribute__((ext_vector_type(8)));
typedef int    v4i     __attribute__((ext_vector_type(4)));

union FragAB { bf16x16 v; uint4 q[2]; };

__device__ __forceinline__ u32 bf16bits(float f) {
    u32 u = __float_as_uint(f);
    return (u + 0x7FFFu + ((u >> 16) & 1u)) >> 16;     // round-to-nearest-even
}
__device__ __forceinline__ u32 pk2(float a, float b) {
    return bf16bits(a) | (bf16bits(b) << 16);
}

__device__ __forceinline__ f32x8 wmma_bf16(bf16x16 a, bf16x16 b, f32x8 c) {
    // D = A(16x32 bf16) x B(32x16 bf16) + C(16x16 f32)
    return __builtin_amdgcn_wmma_f32_16x16x32_bf16(false, a, false, b, (short)0, c, false, false);
}

// ---------------- gfx1250 async global->LDS (ASYNCcnt path), guarded ----------------
#if __has_builtin(__builtin_amdgcn_global_load_async_to_lds_b128)
#define HAVE_ASYNC_LDS 1
typedef __attribute__((address_space(1))) v4i* gp128_t;
typedef __attribute__((address_space(3))) v4i* lp128_t;
__device__ __forceinline__ void async_copy16(const void* g, void* l) {
    __builtin_amdgcn_global_load_async_to_lds_b128((gp128_t)g, (lp128_t)l, 0, 0);
}
__device__ __forceinline__ void wait_async() {
#if __has_builtin(__builtin_amdgcn_s_wait_asynccnt)
    __builtin_amdgcn_s_wait_asynccnt(0);
#else
    asm volatile("s_wait_asynccnt 0x0" ::: "memory");
#endif
}
#else
#define HAVE_ASYNC_LDS 0
__device__ __forceinline__ void async_copy16(const void* g, void* l) {
    *reinterpret_cast<uint4*>(l) = *reinterpret_cast<const uint4*>(g);
}
__device__ __forceinline__ void wait_async() {}
#endif

// A fragment (16x32 bf16): lanes 0-15 rows M, K 0..7 & 16..23; lanes 16-31 K 8..15 & 24..31
__device__ __forceinline__ bf16x16 frag_a(const u32* S, int mi, int lane) {
    int row = mi * 16 + (lane & 15);
    int kb  = (lane & 16) ? 4 : 0;                      // dword offset (2 bf16/dword)
    FragAB f;
    f.q[0] = *reinterpret_cast<const uint4*>(S + row * 20 + kb);
    f.q[1] = *reinterpret_cast<const uint4*>(S + row * 20 + 8 + kb);
    return f.v;
}
// B fragment (32x16 bf16): lane&15 -> column N; lanes 0-15 K 0..7 & 16..23; lanes 16-31 K 8..15 & 24..31
__device__ __forceinline__ bf16x16 frag_b(const u32* S, int ni, int lane) {
    int row = ni * 16 + (lane & 15);
    int kb  = (lane & 16) ? 4 : 0;
    FragAB f;
    f.q[0] = *reinterpret_cast<const uint4*>(S + row * 20 + kb);
    f.q[1] = *reinterpret_cast<const uint4*>(S + row * 20 + 8 + kb);
    return f.v;
}

// =====================================================================
// Router: wave-per-token. logits -> softmax -> top2 -> lists + bf16(x)
// =====================================================================
__global__ __launch_bounds__(256) void moe_router(
    const float* __restrict__ x, const float* __restrict__ gw,
    u32* __restrict__ xb, int* __restrict__ counts,
    int* __restrict__ tei, float* __restrict__ twf,
    float* __restrict__ probs, float* __restrict__ lse2)
{
    int wave = threadIdx.x >> 5, lane = threadIdx.x & 31;
    int token = blockIdx.x * 8 + wave;

    const float2* xp = reinterpret_cast<const float2*>(x + (size_t)token * HID);
    u32* xrow = xb + (size_t)token * (HID / 2);

    float acc[NE];
#pragma unroll
    for (int e = 0; e < NE; ++e) acc[e] = 0.0f;

    for (int j = 0; j < HID / 64; ++j) {
        int idx = lane + 32 * j;
        float2 xv = xp[idx];
        xrow[idx] = pk2(xv.x, xv.y);
#pragma unroll
        for (int e = 0; e < NE; ++e) {
            float2 gv = reinterpret_cast<const float2*>(gw + e * HID)[idx];
            acc[e] += xv.x * gv.x + xv.y * gv.y;
        }
    }
#pragma unroll
    for (int e = 0; e < NE; ++e)
#pragma unroll
        for (int o = 16; o; o >>= 1) acc[e] += __shfl_xor(acc[e], o, 32);

    if (lane == 0) {
        float m = acc[0];
        for (int e = 1; e < NE; ++e) m = fmaxf(m, acc[e]);
        float p[NE], s = 0.0f;
        for (int e = 0; e < NE; ++e) { p[e] = __expf(acc[e] - m); s += p[e]; }
        float inv = 1.0f / s;
        for (int e = 0; e < NE; ++e) { p[e] *= inv; probs[token * NE + e] = p[e]; }
        float lse = m + __logf(s);
        lse2[token] = lse * lse;

        int e1 = 0;
        for (int e = 1; e < NE; ++e) if (p[e] > p[e1]) e1 = e;
        int e2 = (e1 == 0) ? 1 : 0;
        for (int e = 0; e < NE; ++e) if (e != e1 && p[e] > p[e2]) e2 = e;
        float wsum = p[e1] + p[e2];
        tei[token * 2 + 0] = e1; tei[token * 2 + 1] = e2;
        twf[token * 2 + 0] = p[e1] / wsum; twf[token * 2 + 1] = p[e2] / wsum;
        atomicAdd(&counts[e1], 1);
        atomicAdd(&counts[e2], 1);
    }
}

// =====================================================================
// Scan: deterministic loss reductions + padded prefix offsets
// =====================================================================
__global__ __launch_bounds__(256) void moe_scan(
    const int* __restrict__ counts, const float* __restrict__ probs,
    const float* __restrict__ lse2, int* __restrict__ padoff,
    float* __restrict__ losses)
{
    __shared__ float red[256];
    int tid = threadIdx.x;
    float psum[NE];
#pragma unroll
    for (int e = 0; e < NE; ++e) {
        float s = 0.0f;
        for (int t = tid; t < TOK; t += 256) s += probs[t * NE + e];
        red[tid] = s; __syncthreads();
        for (int o = 128; o; o >>= 1) { if (tid < o) red[tid] += red[tid + o]; __syncthreads(); }
        psum[e] = red[0]; __syncthreads();
    }
    float z = 0.0f;
    for (int t = tid; t < TOK; t += 256) z += lse2[t];
    red[tid] = z; __syncthreads();
    for (int o = 128; o; o >>= 1) { if (tid < o) red[tid] += red[tid + o]; __syncthreads(); }
    z = red[0];

    if (tid == 0) {
        int off = 0;
        for (int e = 0; e < NE; ++e) { padoff[e] = off; off += ((counts[e] + 63) >> 6) << 6; }
        padoff[NE] = off;
        float lb = 0.0f;
        for (int e = 0; e < NE; ++e)
            lb += ((float)counts[e] / (float)TOK) * (psum[e] / (float)TOK);
        losses[0] = (float)NE * lb * 0.01f;
        losses[1] = (z / (float)TOK) * 0.001f;
    }
}

// =====================================================================
// Build per-expert compacted (token, weight) lists
// =====================================================================
__global__ __launch_bounds__(256) void moe_build(
    const int* __restrict__ tei, const float* __restrict__ twf,
    const int* __restrict__ padoff, int* __restrict__ cursors,
    int* __restrict__ rowsL, float* __restrict__ wtsL)
{
    int t = blockIdx.x * 256 + threadIdx.x;
    if (t >= TOK) return;
#pragma unroll
    for (int k = 0; k < TOPK; ++k) {
        int e   = tei[t * 2 + k];
        float w = twf[t * 2 + k];
        int p = atomicAdd(&cursors[e], 1);
        int g = padoff[e] + p;
        rowsL[g] = t;
        wtsL[g]  = w;
    }
}

// =====================================================================
// Grouped GEMM 1: Hact = silu(Xg @ Wg^T) * (Xg @ Wu^T)   (bf16 out)
// tile: M=64 tokens x N=128 inter, K-step 32, 8 waves x (2x2) subtiles
// =====================================================================
__global__ __launch_bounds__(256) void moe_gateup(
    const u32* __restrict__ xb, const float* __restrict__ wgate,
    const float* __restrict__ wup, const int* __restrict__ rowsL,
    const int* __restrict__ padoff, unsigned short* __restrict__ hact)
{
    int e    = blockIdx.z;
    int base = padoff[e];
    int cnt  = padoff[e + 1] - base;          // already multiple of 64
    int mbase = blockIdx.x * 64;
    if (mbase >= cnt) return;
    int nbase = blockIdx.y * 128;

    __shared__ u32 As[64 * 20];
    __shared__ u32 Bg[128 * 20];
    __shared__ u32 Bu[128 * 20];
    __shared__ u32 Hs[64 * 64];               // staged 64x128 bf16 output tile

    int tid  = threadIdx.x;
    int lane = tid & 31;
    int wave = tid >> 5;
    int mi0 = (wave >> 2) * 2;
    int ni0 = (wave & 3) * 2;

    f32x8 accg[2][2] = {};
    f32x8 accu[2][2] = {};

    int ar = tid >> 2, ac = tid & 3;                   // A: 64 rows x 4 chunks(16B)
    int atok = rowsL[base + mbase + ar];
    const u32* asrc0 = xb + (size_t)atok * (HID / 2) + ac * 4;
    int br = tid >> 1, bh = tid & 1;                   // B: 128 rows x 2 halves(16 floats)
    size_t gofs = ((size_t)e * INTER + (size_t)(nbase + br)) * HID + (size_t)bh * 16;

    for (int kt = 0; kt < HID / 32; ++kt) {
        __syncthreads();
        // A tile: gathered bf16 rows -> LDS via async DMA (16B/thread)
        async_copy16(asrc0 + kt * 16, &As[ar * 20 + ac * 4]);
        // prefetch next K-step weight lines into cache
        if (kt + 1 < HID / 32) {
            __builtin_prefetch((const void*)(wgate + gofs + (size_t)(kt + 1) * 32), 0, 1);
            __builtin_prefetch((const void*)(wup   + gofs + (size_t)(kt + 1) * 32), 0, 1);
        }
        // B tiles: f32 -> bf16 convert in flight
        const float4* pg = reinterpret_cast<const float4*>(wgate + gofs + (size_t)kt * 32);
        const float4* pu = reinterpret_cast<const float4*>(wup   + gofs + (size_t)kt * 32);
#pragma unroll
        for (int f = 0; f < 4; ++f) {
            float4 vg = pg[f];
            float4 vu = pu[f];
            uint2 qg = make_uint2(pk2(vg.x, vg.y), pk2(vg.z, vg.w));
            uint2 qu = make_uint2(pk2(vu.x, vu.y), pk2(vu.z, vu.w));
            *reinterpret_cast<uint2*>(&Bg[br * 20 + bh * 8 + f * 2]) = qg;
            *reinterpret_cast<uint2*>(&Bu[br * 20 + bh * 8 + f * 2]) = qu;
        }
        wait_async();
        __syncthreads();

        bf16x16 a0  = frag_a(As, mi0,     lane);
        bf16x16 a1  = frag_a(As, mi0 + 1, lane);
        bf16x16 g0  = frag_b(Bg, ni0,     lane);
        bf16x16 g1  = frag_b(Bg, ni0 + 1, lane);
        bf16x16 u0  = frag_b(Bu, ni0,     lane);
        bf16x16 u1  = frag_b(Bu, ni0 + 1, lane);

        accg[0][0] = wmma_bf16(a0, g0, accg[0][0]);
        accg[0][1] = wmma_bf16(a0, g1, accg[0][1]);
        accg[1][0] = wmma_bf16(a1, g0, accg[1][0]);
        accg[1][1] = wmma_bf16(a1, g1, accg[1][1]);
        accu[0][0] = wmma_bf16(a0, u0, accu[0][0]);
        accu[0][1] = wmma_bf16(a0, u1, accu[0][1]);
        accu[1][0] = wmma_bf16(a1, u0, accu[1][0]);
        accu[1][1] = wmma_bf16(a1, u1, accu[1][1]);
    }

    // fused SiLU(g)*u epilogue -> staged bf16 tile in LDS (fast v_rcp_f32 path)
    unsigned short* Hs16 = reinterpret_cast<unsigned short*>(Hs);
    int mh = (lane >> 4) << 3;
    int nn = lane & 15;
#pragma unroll
    for (int i = 0; i < 2; ++i) {
#pragma unroll
        for (int r = 0; r < 8; ++r) {
            int m = (mi0 + i) * 16 + r + mh;
#pragma unroll
            for (int j = 0; j < 2; ++j) {
                float g = accg[i][j][r];
                float u = accu[i][j][r];
                float h = g * __builtin_amdgcn_rcpf(1.0f + __expf(-g)) * u;
                Hs16[m * 128 + (ni0 + j) * 16 + nn] = (unsigned short)bf16bits(h);
            }
        }
    }
    __syncthreads();
    // cooperative coalesced b128 stores of the staged tile
#pragma unroll
    for (int ii = 0; ii < 4; ++ii) {
        int chunk = tid + 256 * ii;            // 0..1023
        int row = chunk >> 4;
        int c   = chunk & 15;
        *reinterpret_cast<uint4*>(hact + (size_t)(base + mbase + row) * INTER + nbase + c * 8) =
            *reinterpret_cast<const uint4*>(Hs + row * 64 + c * 4);
    }
}

// =====================================================================
// Grouped GEMM 2: out += w_t * (Hact @ Wd^T), f32 atomic combine
// tile: M=64 rows x N=128 hidden, K over 4096
// =====================================================================
__global__ __launch_bounds__(256) void moe_down(
    const u32* __restrict__ hact, const float* __restrict__ wdown,
    const int* __restrict__ rowsL, const float* __restrict__ wtsL,
    const int* __restrict__ padoff, float* __restrict__ out)
{
    int e    = blockIdx.z;
    int base = padoff[e];
    int cnt  = padoff[e + 1] - base;
    int mbase = blockIdx.x * 64;
    if (mbase >= cnt) return;
    int nbase = blockIdx.y * 128;

    __shared__ u32 As[64 * 20];
    __shared__ u32 Bd[128 * 20];

    int tid  = threadIdx.x;
    int lane = tid & 31;
    int wave = tid >> 5;
    int mi0 = (wave >> 2) * 2;
    int ni0 = (wave & 3) * 2;

    f32x8 acc[2][2] = {};

    int ar = tid >> 2, ac = tid & 3;
    const u32* asrc0 = hact + (size_t)(base + mbase + ar) * (INTER / 2) + ac * 4;
    int br = tid >> 1, bh = tid & 1;
    size_t gofs = ((size_t)e * HID + (size_t)(nbase + br)) * INTER + (size_t)bh * 16;

    for (int kt = 0; kt < INTER / 32; ++kt) {
        __syncthreads();
        async_copy16(asrc0 + kt * 16, &As[ar * 20 + ac * 4]);
        if (kt + 1 < INTER / 32) {
            __builtin_prefetch((const void*)(wdown + gofs + (size_t)(kt + 1) * 32), 0, 1);
        }
        const float4* pd = reinterpret_cast<const float4*>(wdown + gofs + (size_t)kt * 32);
#pragma unroll
        for (int f = 0; f < 4; ++f) {
            float4 vd = pd[f];
            *reinterpret_cast<uint2*>(&Bd[br * 20 + bh * 8 + f * 2]) =
                make_uint2(pk2(vd.x, vd.y), pk2(vd.z, vd.w));
        }
        wait_async();
        __syncthreads();

        bf16x16 a0 = frag_a(As, mi0,     lane);
        bf16x16 a1 = frag_a(As, mi0 + 1, lane);
        bf16x16 b0 = frag_b(Bd, ni0,     lane);
        bf16x16 b1 = frag_b(Bd, ni0 + 1, lane);

        acc[0][0] = wmma_bf16(a0, b0, acc[0][0]);
        acc[0][1] = wmma_bf16(a0, b1, acc[0][1]);
        acc[1][0] = wmma_bf16(a1, b0, acc[1][0]);
        acc[1][1] = wmma_bf16(a1, b1, acc[1][1]);
    }

    int mh = (lane >> 4) << 3;
    int nn = lane & 15;
#pragma unroll
    for (int i = 0; i < 2; ++i) {
#pragma unroll
        for (int r = 0; r < 8; ++r) {
            int m   = (mi0 + i) * 16 + r + mh;
            int gid = base + mbase + m;
            int token = rowsL[gid];
            float w   = wtsL[gid];
            size_t orow = (size_t)token * HID + nbase;
#pragma unroll
            for (int j = 0; j < 2; ++j) {
                atomicAdd(out + orow + (ni0 + j) * 16 + nn, w * acc[i][j][r]);
            }
        }
    }
}

// =====================================================================
extern "C" void kernel_launch(void* const* d_in, const int* in_sizes, int n_in,
                              void* d_out, int out_size, void* d_ws, size_t ws_size,
                              hipStream_t stream)
{
    (void)in_sizes; (void)n_in; (void)out_size; (void)ws_size;
    const float* x     = (const float*)d_in[0];
    const float* gw    = (const float*)d_in[1];
    const float* wgate = (const float*)d_in[2];
    const float* wup   = (const float*)d_in[3];
    const float* wdown = (const float*)d_in[4];
    float* out = (float*)d_out;

    char* ws = (char*)d_ws;
    int*   counts = (int*)(ws + OFF_COUNTS);
    int*   curs   = (int*)(ws + OFF_CURS);
    int*   padoff = (int*)(ws + OFF_PADOFF);
    int*   tei    = (int*)(ws + OFF_TEI);
    float* twf    = (float*)(ws + OFF_TWF);
    int*   rowsL  = (int*)(ws + OFF_ROWS);
    float* wtsL   = (float*)(ws + OFF_WTS);
    float* probs  = (float*)(ws + OFF_PROBS);
    float* lse2   = (float*)(ws + OFF_LSE);
    u32*   xb     = (u32*)(ws + OFF_XB);
    unsigned short* hact = (unsigned short*)(ws + OFF_HACT);

    // zero output accumulator and counters/lists (graph-capturable)
    (void)hipMemsetAsync(out, 0, (size_t)TOK * HID * sizeof(float), stream);
    (void)hipMemsetAsync(ws, 0, (size_t)OFF_PROBS, stream);

    moe_router<<<dim3(TOK / 8), dim3(256), 0, stream>>>(
        x, gw, xb, counts, tei, twf, probs, lse2);

    moe_scan<<<dim3(1), dim3(256), 0, stream>>>(
        counts, probs, lse2, padoff, out + (size_t)TOK * HID);

    moe_build<<<dim3(TOK / 256), dim3(256), 0, stream>>>(
        tei, twf, padoff, curs, rowsL, wtsL);

    moe_gateup<<<dim3(128, INTER / 128, NE), dim3(256), 0, stream>>>(
        xb, wgate, wup, rowsL, padoff, hact);

    moe_down<<<dim3(128, HID / 128, NE), dim3(256), 0, stream>>>(
        (const u32*)hact, wdown, rowsL, wtsL, padoff, out);
}
